// EvolveGCN_75239237091506
// MI455X (gfx1250) — compile-verified, matrix operand order checked by reading, not run on hardware
//
#include <hip/hip_runtime.h>
#include <math.h>

// Problem constants (match reference)
constexpr int N = 100000;
constexpr int D = 128;
constexpr int E = 3200000;

typedef __attribute__((ext_vector_type(16))) __bf16 v16bf;
typedef __attribute__((ext_vector_type(8)))  __bf16 v8bf;
typedef __attribute__((ext_vector_type(8)))  float  v8f;

__device__ __forceinline__ unsigned short f2bf(float f) {
    unsigned int u = __builtin_bit_cast(unsigned int, f);
    unsigned int r = u + 0x7FFFu + ((u >> 16) & 1u);   // round-to-nearest-even
    return (unsigned short)(r >> 16);
}
__device__ __forceinline__ float sigm(float x) { return 1.0f / (1.0f + __expf(-x)); }

// ---------------- utility zero kernels ----------------
__global__ void zero_f32(float* p, long long n) {
    long long i = (long long)blockIdx.x * blockDim.x + threadIdx.x;
    if (i < n) p[i] = 0.0f;
}
__global__ void zero_i32(int* p, long long n) {
    long long i = (long long)blockIdx.x * blockDim.x + threadIdx.x;
    if (i < n) p[i] = 0;
}

// ---------------- ||p||^2 (D=128, one block) ----------------
__global__ void pnorm2_kernel(const float* __restrict__ p, float* __restrict__ out) {
    __shared__ float s[128];
    float v = p[threadIdx.x];
    s[threadIdx.x] = v * v;
    __syncthreads();
    for (int k = 64; k > 0; k >>= 1) {
        if (threadIdx.x < k) s[threadIdx.x] += s[threadIdx.x + k];
        __syncthreads();
    }
    if (threadIdx.x == 0) out[0] = s[0];
}

// ---------------- score[i] = dot(x[i], p) / ||p|| ; one wave per row ----------------
__global__ __launch_bounds__(256) void score_kernel(const float* __restrict__ x,
                                                    const float* __restrict__ p,
                                                    const float* __restrict__ pnorm2,
                                                    float* __restrict__ score) {
    const int lane = threadIdx.x & 31;
    const int row  = blockIdx.x * 8 + (threadIdx.x >> 5);
    float acc = 0.0f;
    #pragma unroll
    for (int j = 0; j < 4; ++j) {
        int d = lane + 32 * j;
        acc += x[(size_t)row * D + d] * p[d];
    }
    #pragma unroll
    for (int m = 16; m >= 1; m >>= 1) acc += __shfl_xor(acc, m, 32);
    if (lane == 0) score[row] = acc * rsqrtf(pnorm2[0]);
}

// ---------------- deterministic 128-pass argmax top-k (single block) ----------------
__global__ __launch_bounds__(256) void topk_kernel(const float* __restrict__ score,
                                                   int* __restrict__ taken,
                                                   int* __restrict__ perm,
                                                   float* __restrict__ tscore) {
    __shared__ float sv[256];
    __shared__ int   si[256];
    const int tid = threadIdx.x;
    for (int iter = 0; iter < D; ++iter) {
        float best = -3.4e38f; int bi = 0x7FFFFFFF;
        for (int i = tid; i < N; i += 256) {
            if (!taken[i]) {
                float s = score[i];
                if (s > best) { best = s; bi = i; }   // first occurrence wins
            }
        }
        sv[tid] = best; si[tid] = bi;
        __syncthreads();
        for (int k = 128; k > 0; k >>= 1) {
            if (tid < k) {
                bool t = (sv[tid + k] > sv[tid]) ||
                         (sv[tid + k] == sv[tid] && si[tid + k] < si[tid]);
                if (t) { sv[tid] = sv[tid + k]; si[tid] = si[tid + k]; }
            }
            __syncthreads();
        }
        if (tid == 0) { perm[iter] = si[0]; tscore[iter] = sv[0]; taken[si[0]] = 1; }
        __syncthreads();
    }
}

// ---------------- x_tilde[i,j] = x[perm[i], j] * tanh(top_score[i]) ----------------
__global__ void xtilde_kernel(const float* __restrict__ x,
                              const int* __restrict__ perm,
                              const float* __restrict__ tscore,
                              float* __restrict__ xt) {
    int idx = blockIdx.x * blockDim.x + threadIdx.x;   // 0..16383
    int i = idx >> 7, j = idx & 127;
    xt[idx] = x[(size_t)perm[i] * D + j] * tanhf(tscore[i]);
}

// ---------------- GRU gate pre-activations: gi = xt @ W_ih^T + b_ih ; gh = W0 @ W_hh^T + b_hh ----------------
__global__ __launch_bounds__(256) void gru_mm_kernel(const float* __restrict__ xt,
                                                     const float* __restrict__ W0,
                                                     const float* __restrict__ W_ih,
                                                     const float* __restrict__ W_hh,
                                                     const float* __restrict__ b_ih,
                                                     const float* __restrict__ b_hh,
                                                     float* __restrict__ gi,
                                                     float* __restrict__ gh) {
    int idx = blockIdx.x * blockDim.x + threadIdx.x;   // 0..49151
    int i = idx / 384, g = idx % 384;
    float a = b_ih[g], b = b_hh[g];
    #pragma unroll 8
    for (int k = 0; k < D; ++k) {
        a += xt[i * D + k] * W_ih[g * D + k];
        b += W0[i * D + k] * W_hh[g * D + k];
    }
    gi[idx] = a; gh[idx] = b;
}

// ---------------- GRU combine -> W[i,j] ----------------
__global__ void gru_gate_kernel(const float* __restrict__ gi,
                                const float* __restrict__ gh,
                                const float* __restrict__ W0,
                                float* __restrict__ W) {
    int idx = blockIdx.x * blockDim.x + threadIdx.x;   // 0..16383
    int i = idx >> 7, j = idx & 127;
    const float* gri = gi + (size_t)i * 384;
    const float* grh = gh + (size_t)i * 384;
    float r = sigm(gri[j] + grh[j]);
    float z = sigm(gri[128 + j] + grh[128 + j]);
    float n = tanhf(gri[256 + j] + r * grh[256 + j]);
    W[idx] = (1.0f - z) * n + z * W0[idx];
}

// ---------------- W (k-major) -> Wt bf16 (n-major) ----------------
__global__ void convw_kernel(const float* __restrict__ W, unsigned short* __restrict__ wt) {
    int idx = blockIdx.x * blockDim.x + threadIdx.x;   // 0..16383
    int n = idx >> 7, k = idx & 127;
    wt[n * D + k] = f2bf(W[k * D + n]);
}

// ---------------- x f32 -> bf16 ----------------
__global__ void convx_kernel(const float* __restrict__ x, unsigned short* __restrict__ xb) {
    long long i = (long long)blockIdx.x * blockDim.x + threadIdx.x;
    if (i < (long long)N * D) xb[i] = f2bf(x[i]);
}

// ---------------- xw = x @ W : 16x16 tile per wave, K=128 via 4x WMMA bf16 ----------------
__global__ __launch_bounds__(256) void gemm_xw_kernel(const unsigned short* __restrict__ xb,
                                                      const unsigned short* __restrict__ wt,
                                                      float* __restrict__ xw) {
    const int lane = threadIdx.x & 31;
    const int wv   = threadIdx.x >> 5;      // 0..7 -> column tile
    const int tm   = blockIdx.x;            // row tile (N/16 = 6250)
    const int rowA = (tm << 4) + (lane & 15);
    const int colB = (wv << 4) + (lane & 15);
    const int hi   = (lane >> 4) << 3;      // upper half-wave: K offset +8
    v8f c = {};
    #pragma unroll
    for (int kc = 0; kc < 4; ++kc) {
        const int kb = kc * 32 + hi;
        const v8bf* pa = (const v8bf*)(xb + (size_t)rowA * D + kb);
        const v8bf* pb = (const v8bf*)(wt + (size_t)colB * D + kb);
        v8bf a0 = pa[0], a1 = pa[2];        // +16 elements
        v8bf b0 = pb[0], b1 = pb[2];
        v16bf A, B;
        #pragma unroll
        for (int i = 0; i < 8; ++i) { A[i] = a0[i]; A[8 + i] = a1[i];
                                      B[i] = b0[i]; B[8 + i] = b1[i]; }
        c = __builtin_amdgcn_wmma_f32_16x16x32_bf16(false, A, false, B, (short)0, c,
                                                    false, false);
    }
    const int mb  = (tm << 4) + ((lane >> 4) << 3);
    const int col = (wv << 4) + (lane & 15);
    #pragma unroll
    for (int r = 0; r < 8; ++r) xw[(size_t)(mb + r) * D + col] = c[r];
}

// ---------------- degree via atomics ----------------
__global__ void deg_kernel(const int* __restrict__ col, float* __restrict__ deg) {
    int e = blockIdx.x * blockDim.x + threadIdx.x;
    if (e < E) atomicAdd(&deg[col[e]], 1.0f);
}
__global__ void dinv_kernel(const float* __restrict__ deg, float* __restrict__ dinv) {
    int i = blockIdx.x * blockDim.x + threadIdx.x;
    if (i < N) dinv[i] = rsqrtf(deg[i] + 1.0f);
}

// ---------------- edge scatter: one wave per edge, 4 features per lane ----------------
__global__ __launch_bounds__(256) void aggregate_kernel(const int* __restrict__ row,
                                                        const int* __restrict__ col,
                                                        const float* __restrict__ dinv,
                                                        const float* __restrict__ xw,
                                                        float* __restrict__ agg) {
    const int lane = threadIdx.x & 31;
    const int e = blockIdx.x * 8 + (threadIdx.x >> 5);  // E/8 blocks exactly
    const int r = row[e], cN = col[e];
    const float coef = dinv[r] * dinv[cN];
    const float* src = xw + (size_t)r * D;
    float* dst = agg + (size_t)cN * D;
    #pragma unroll
    for (int j = 0; j < 4; ++j) {
        int d = lane + 32 * j;
        atomicAdd(&dst[d], src[d] * coef);
    }
}

// ---------------- out = rownorm(relu(agg + xw*dinv^2)) ; one wave per node ----------------
__global__ __launch_bounds__(256) void finalize_kernel(const float* __restrict__ agg,
                                                       const float* __restrict__ xw,
                                                       const float* __restrict__ dinv,
                                                       float* __restrict__ out) {
    const int lane = threadIdx.x & 31;
    const int n = blockIdx.x * 8 + (threadIdx.x >> 5);  // N/8 blocks exactly
    const float di = dinv[n];
    const float selfc = di * di;
    float v[4]; float ss = 0.0f;
    #pragma unroll
    for (int j = 0; j < 4; ++j) {
        int d = lane + 32 * j;
        float t = agg[(size_t)n * D + d] + xw[(size_t)n * D + d] * selfc;
        t = t > 0.0f ? t : 0.0f;
        v[j] = t; ss += t * t;
    }
    #pragma unroll
    for (int m = 16; m >= 1; m >>= 1) ss += __shfl_xor(ss, m, 32);
    const float inv = 1.0f / fmaxf(sqrtf(ss), 1e-12f);
    #pragma unroll
    for (int j = 0; j < 4; ++j) {
        int d = lane + 32 * j;
        out[(size_t)n * D + d] = v[j] * inv;
    }
}

extern "C" void kernel_launch(void* const* d_in, const int* in_sizes, int n_in,
                              void* d_out, int out_size, void* d_ws, size_t ws_size,
                              hipStream_t stream) {
    const float* x     = (const float*)d_in[0];         // H_K_prev[0] = first N*D floats
    const int*   edges = (const int*)d_in[1];           // edgelists[0] = first 2*E ints
    const int*   row   = edges;
    const int*   col   = edges + E;
    const float* p     = (const float*)d_in[2];
    const float* W_ih  = (const float*)d_in[3];
    const float* W_hh  = (const float*)d_in[4];
    const float* b_ih  = (const float*)d_in[5];
    const float* b_hh  = (const float*)d_in[6];
    const float* W0    = (const float*)d_in[7];
    float* out = (float*)d_out;

    // workspace carve-up (256B aligned)
    char* base = (char*)d_ws;
    size_t off = 0;
    auto alloc = [&](size_t bytes) -> char* {
        char* r = base + off;
        off = (off + bytes + 255) & ~(size_t)255;
        return r;
    };
    float* pn2    = (float*)alloc(4);
    float* score  = (float*)alloc((size_t)N * 4);
    int*   taken  = (int*)  alloc((size_t)N * 4);
    int*   perm   = (int*)  alloc(128 * 4);
    float* tsc    = (float*)alloc(128 * 4);
    float* xt     = (float*)alloc((size_t)D * D * 4);
    float* gi     = (float*)alloc((size_t)D * 384 * 4);
    float* gh     = (float*)alloc((size_t)D * 384 * 4);
    float* Wm     = (float*)alloc((size_t)D * D * 4);
    unsigned short* wtb = (unsigned short*)alloc((size_t)D * D * 2);
    unsigned short* xb  = (unsigned short*)alloc((size_t)N * D * 2);
    float* xw     = (float*)alloc((size_t)N * D * 4);
    float* deg    = (float*)alloc((size_t)N * 4);
    float* dinv   = (float*)alloc((size_t)N * 4);
    float* agg    = (float*)alloc((size_t)N * D * 4);

    // 1) top-k score path
    pnorm2_kernel<<<1, 128, 0, stream>>>(p, pn2);
    zero_i32<<<(N + 255) / 256, 256, 0, stream>>>(taken, N);
    score_kernel<<<N / 8, 256, 0, stream>>>(x, p, pn2, score);
    topk_kernel<<<1, 256, 0, stream>>>(score, taken, perm, tsc);
    xtilde_kernel<<<(D * D) / 256, 256, 0, stream>>>(x, perm, tsc, xt);

    // 2) GRU -> evolved weight W
    gru_mm_kernel<<<(D * 384) / 256, 256, 0, stream>>>(xt, W0, W_ih, W_hh, b_ih, b_hh, gi, gh);
    gru_gate_kernel<<<(D * D) / 256, 256, 0, stream>>>(gi, gh, W0, Wm);

    // 3) bf16 conversions + WMMA GEMM xw = x @ W
    convw_kernel<<<(D * D) / 256, 256, 0, stream>>>(Wm, wtb);
    convx_kernel<<<(N * D) / 256, 256, 0, stream>>>(x, xb);
    gemm_xw_kernel<<<N / 16, 256, 0, stream>>>(xb, wtb, xw);

    // 4) GCN normalization + edge aggregation
    zero_f32<<<(N + 255) / 256, 256, 0, stream>>>(deg, N);
    deg_kernel<<<(E + 255) / 256, 256, 0, stream>>>(col, deg);
    dinv_kernel<<<(N + 255) / 256, 256, 0, stream>>>(deg, dinv);
    zero_f32<<<((long long)N * D + 255) / 256, 256, 0, stream>>>(agg, (long long)N * D);
    aggregate_kernel<<<E / 8, 256, 0, stream>>>(row, col, dinv, xw, agg);

    // 5) relu + row L2 normalize
    finalize_kernel<<<N / 8, 256, 0, stream>>>(agg, xw, dinv, out);
}
